// DSMoE_9715216024107
// MI455X (gfx1250) — matrix operand order repaired and checked by reading, not compiled, not run on hardware
//
#include <hip/hip_runtime.h>
#include <hip/hip_bf16.h>

typedef __attribute__((ext_vector_type(16))) _Float16 v16h;
typedef __attribute__((ext_vector_type(8)))  _Float16 v8h;
typedef __attribute__((ext_vector_type(4)))  _Float16 v4h;
typedef __attribute__((ext_vector_type(8)))  float    v8f;
typedef int v4i_vs __attribute__((vector_size(16)));   // b128 payload type

#define NTOK  8192
#define CDIM  256
#define EXPN  32
#define HDIM  1024
#define TM    64          // tokens per tile
#define HC    128         // H-chunk
#define NEG_INF (-3.0e38f)

// workspace layout (bytes)
#define OFF_CNT  0
#define OFF_TL   1024
#define OFF_WL   (OFF_TL + (size_t)EXPN * NTOK * 4)
#define OFF_X16  (OFF_WL + (size_t)EXPN * NTOK * 4)
#define OFF_WFC  (OFF_X16 + (size_t)NTOK * CDIM * 2)
#define OFF_WPJ  (OFF_WFC + (size_t)EXPN * CDIM * HDIM * 2)
#define WS_NEED  (OFF_WPJ + (size_t)EXPN * HDIM * CDIM * 2)

#if defined(__gfx1250__) && \
    __has_builtin(__builtin_amdgcn_global_load_async_to_lds_b128) && \
    __has_builtin(__builtin_amdgcn_s_wait_asynccnt)
#define USE_ASYNC_LDS 1
#define AS1 __attribute__((address_space(1)))
#define AS3 __attribute__((address_space(3)))
#endif

// ---------------------------------------------------------------------------
// Kernel 1: zero output accumulator, reset per-expert counters.
// ---------------------------------------------------------------------------
__global__ void moe_init(float* __restrict__ out, int* __restrict__ cnt) {
    int i = blockIdx.x * blockDim.x + threadIdx.x;
    if (i < NTOK * CDIM) out[i] = 0.0f;
    if (i < EXPN) cnt[i] = (i == 0) ? NTOK : 0;
}

// ---------------------------------------------------------------------------
// x f32 -> f16 (vectorized by 4)
// ---------------------------------------------------------------------------
__global__ void cvt_x(const float* __restrict__ src, _Float16* __restrict__ dst) {
    int i = (blockIdx.x * blockDim.x + threadIdx.x) * 4;
    float4 f = *(const float4*)(src + i);
    v4h h;
    h[0] = (_Float16)f.x; h[1] = (_Float16)f.y;
    h[2] = (_Float16)f.z; h[3] = (_Float16)f.w;
    *(v4h*)(dst + i) = h;
}

// ---------------------------------------------------------------------------
// per-expert transpose + f32->f16: src [E][R][Cc] f32 -> dst [E][Cc][R] f16
// ---------------------------------------------------------------------------
__global__ void transpose_cvt(const float* __restrict__ src, _Float16* __restrict__ dst,
                              int R, int Cc) {
    __shared__ float tile[32][33];
    const float* s = src + (size_t)blockIdx.z * R * Cc;
    _Float16*    d = dst + (size_t)blockIdx.z * R * Cc;
    int c0 = blockIdx.x * 32, r0 = blockIdx.y * 32;
    int tx = threadIdx.x, ty = threadIdx.y;      // 32 x 8
    #pragma unroll
    for (int i = 0; i < 32; i += 8)
        tile[ty + i][tx] = s[(size_t)(r0 + ty + i) * Cc + c0 + tx];
    __syncthreads();
    #pragma unroll
    for (int i = 0; i < 32; i += 8)
        d[(size_t)(c0 + ty + i) * R + r0 + tx] = (_Float16)tile[tx][ty + i];
}

// ---------------------------------------------------------------------------
// Kernel 2: gating. One wave32 per token. lane j (j<31) owns routed expert j.
// ---------------------------------------------------------------------------
__global__ void moe_gate(const float* __restrict__ x,
                         const float* __restrict__ Wg,
                         const float* __restrict__ ebias,
                         float* __restrict__ rw,
                         int* __restrict__ cnt,
                         int* __restrict__ tlist,
                         float* __restrict__ wlist) {
    const int wave = threadIdx.x >> 5;
    const int lane = threadIdx.x & 31;
    const int n = blockIdx.x * 8 + wave;
    if (n >= NTOK) return;

    float acc = 0.0f;
    if (lane < 31) {
        const float* xr = x + (size_t)n * CDIM;
        #pragma unroll 4
        for (int k = 0; k < CDIM; ++k) acc += xr[k] * Wg[k * 31 + lane];
    }
    float logit = (lane < 31) ? acc : NEG_INF;

    float m = logit;
    for (int off = 16; off; off >>= 1) m = fmaxf(m, __shfl_xor(m, off));
    float p = (lane < 31) ? __expf(logit - m) : 0.0f;
    float s = p;
    for (int off = 16; off; off >>= 1) s += __shfl_xor(s, off);
    float gate = p / s;
    float biased = (lane < 31) ? (gate + ebias[lane]) : NEG_INF;

    // top-3 by repeated argmax (value desc, index asc on ties)
    float cur = biased;
    bool  sel = false;
    float sumtop = 0.0f;
    for (int r = 0; r < 3; ++r) {
        float v = cur; int bi = lane;
        for (int off = 16; off; off >>= 1) {
            float ov = __shfl_xor(v, off);
            int   oi = __shfl_xor(bi, off);
            if (ov > v || (ov == v && oi < bi)) { v = ov; bi = oi; }
        }
        sumtop += v;
        if (lane == bi) { sel = true; cur = NEG_INF; }
    }
    float wfinal = sel ? (biased / sumtop) * 0.75f : 0.0f;

    float* rwn = rw + (size_t)n * EXPN;
    if (lane < 31)  rwn[lane + 1] = wfinal;
    if (lane == 31) rwn[0] = 0.25f;

    if (lane == 0) { tlist[n] = n; wlist[n] = 0.25f; }
    if (sel) {
        int e = lane + 1;
        int pos = atomicAdd(&cnt[e], 1);
        tlist[(size_t)e * NTOK + pos] = n;
        wlist[(size_t)e * NTOK + pos] = wfinal;
    }
}

// ---------------------------------------------------------------------------
// Kernel 3 (fast): pre-transposed f16 weights; B-fragments loaded straight
// from global (L2-resident), A via LDS; x gather via async-to-LDS.
// ---------------------------------------------------------------------------
__global__ __launch_bounds__(256)
void moe_expert_fast(const _Float16* __restrict__ x16,
                     const _Float16* __restrict__ WfcT,   // [E][H][C]
                     const _Float16* __restrict__ WpjT,   // [E][C][H]
                     const int*   __restrict__ cnt,
                     const int*   __restrict__ tlist,
                     const float* __restrict__ wlist,
                     float* __restrict__ out) {
    __shared__ _Float16 aS[TM * CDIM];   // 32 KB gathered tokens (f16)
    __shared__ _Float16 hS[TM * HC];     // 16 KB relu^2 chunk (f16)

    const int e = blockIdx.y;
    const int tile0 = blockIdx.x * TM;
    const int cntE = cnt[e];
    if (tile0 >= cntE) return;

    const int tid  = threadIdx.x;
    const int lane = tid & 31;
    const int w    = tid >> 5;
    const int col  = lane & 15;
    const int hiH  = lane >> 4;
    const int koff = hiH * 16;
    const int mofs = hiH * 8;

    const _Float16* Wfe = WfcT + (size_t)e * HDIM * CDIM;
    const _Float16* Wpe = WpjT + (size_t)e * CDIM * HDIM;
    const int*   tl = tlist + (size_t)e * NTOK;
    const float* wl = wlist + (size_t)e * NTOK;

    // ---- gather x16 rows into aS ----
    {
        int row  = tid >> 2;
        int cb   = (tid & 3) * 64;        // 64 halves = 128 B per thread
        int gRow = tile0 + row;
        int tok  = (gRow < cntE) ? tl[gRow] : 0;
        const _Float16* src = x16 + (size_t)tok * CDIM + cb;
        _Float16* dstp = aS + row * CDIM + cb;
#ifdef USE_ASYNC_LDS
        #pragma unroll
        for (int i = 0; i < 64; i += 8)
            __builtin_amdgcn_global_load_async_to_lds_b128(
                (AS1 v4i_vs*)(src + i), (AS3 v4i_vs*)(dstp + i), 0, 0);
        __builtin_amdgcn_s_wait_asynccnt(0);
#else
        #pragma unroll
        for (int i = 0; i < 64; i += 8)
            *(v8h*)(dstp + i) = *(const v8h*)(src + i);
#endif
    }
    __syncthreads();

    v8f oacc[4][2];
    #pragma unroll
    for (int mm = 0; mm < 4; ++mm) { oacc[mm][0] = (v8f)0.0f; oacc[mm][1] = (v8f)0.0f; }

    for (int hc = 0; hc < HDIM / HC; ++hc) {
        const int hb = hc * HC;
        v8f hacc[4];
        #pragma unroll
        for (int mm = 0; mm < 4; ++mm) hacc[mm] = (v8f)0.0f;

        // ---- GEMM1: h[64x128] = aS @ Wfc[:, hb:hb+128] (barrier-free loop) ----
        for (int k0 = 0; k0 < CDIM; k0 += 32) {
            v16h bfrag = *(const v16h*)(Wfe + (size_t)(hb + 16 * w + col) * CDIM + k0 + hiH * 16);
            #pragma unroll
            for (int mm = 0; mm < 4; ++mm) {
                const _Float16* ap = aS + (16 * mm + col) * CDIM + k0 + koff;
                v16h afrag;
                ((v8h*)&afrag)[0] = *(const v8h*)ap;
                ((v8h*)&afrag)[1] = *(const v8h*)(ap + 8);
                hacc[mm] = __builtin_amdgcn_wmma_f32_16x16x32_f16(
                    false, afrag, false, bfrag, (short)0, hacc[mm], false, false);
            }
        }

        __syncthreads();   // previous chunk's hS readers done
        #pragma unroll
        for (int mm = 0; mm < 4; ++mm)
            #pragma unroll
            for (int r = 0; r < 8; ++r) {
                float v = fmaxf(hacc[mm][r], 0.0f);
                v *= v;
                hS[(16 * mm + mofs + r) * HC + 16 * w + col] = (_Float16)v;
            }
        __syncthreads();

        // ---- GEMM2: out += relu2(h) @ Wpj[hb:hb+128, :] (barrier-free loop) ----
        for (int k0 = 0; k0 < HC; k0 += 32) {
            v16h b0 = *(const v16h*)(Wpe + (size_t)(32 * w      + col) * HDIM + hb + k0 + hiH * 16);
            v16h b1 = *(const v16h*)(Wpe + (size_t)(32 * w + 16 + col) * HDIM + hb + k0 + hiH * 16);
            #pragma unroll
            for (int mm = 0; mm < 4; ++mm) {
                const _Float16* ap = hS + (16 * mm + col) * HC + k0 + koff;
                v16h afrag;
                ((v8h*)&afrag)[0] = *(const v8h*)ap;
                ((v8h*)&afrag)[1] = *(const v8h*)(ap + 8);
                oacc[mm][0] = __builtin_amdgcn_wmma_f32_16x16x32_f16(
                    false, afrag, false, b0, (short)0, oacc[mm][0], false, false);
                oacc[mm][1] = __builtin_amdgcn_wmma_f32_16x16x32_f16(
                    false, afrag, false, b1, (short)0, oacc[mm][1], false, false);
            }
        }
    }

    // ---- epilogue: scale by router weight, scatter-add into out ----
    #pragma unroll
    for (int mm = 0; mm < 4; ++mm) {
        int Mbase = 16 * mm + mofs;
        #pragma unroll
        for (int r = 0; r < 8; ++r) {
            int gRow = tile0 + Mbase + r;
            if (gRow >= cntE) continue;
            int   tok = tl[gRow];
            float wt  = wl[gRow];
            float* orow = out + (size_t)tok * CDIM;
            atomicAdd(orow + 32 * w + col,      oacc[mm][0][r] * wt);
            atomicAdd(orow + 32 * w + 16 + col, oacc[mm][1][r] * wt);
        }
    }
}

// ---------------------------------------------------------------------------
// Kernel 3 (fallback, small ws): convert f32 weights in-kernel via LDS.
// ---------------------------------------------------------------------------
__global__ __launch_bounds__(256)
void moe_expert_fb(const float* __restrict__ x,
                   const float* __restrict__ Wfc,
                   const float* __restrict__ Wpj,
                   const int*   __restrict__ cnt,
                   const int*   __restrict__ tlist,
                   const float* __restrict__ wlist,
                   float* __restrict__ out) {
    __shared__ _Float16 aS[TM * CDIM];
    __shared__ _Float16 hS[TM * HC];
    __shared__ _Float16 bT[256 * 32];

    const int e = blockIdx.y;
    const int tile0 = blockIdx.x * TM;
    const int cntE = cnt[e];
    if (tile0 >= cntE) return;

    const int tid  = threadIdx.x;
    const int lane = tid & 31;
    const int w    = tid >> 5;
    const int col  = lane & 15;
    const int hiH  = lane >> 4;
    const int koff = hiH * 16;
    const int mofs = hiH * 8;

    const float* Wfe = Wfc + (size_t)e * CDIM * HDIM;
    const float* Wpe = Wpj + (size_t)e * HDIM * CDIM;
    const int*   tl  = tlist + (size_t)e * NTOK;
    const float* wl  = wlist + (size_t)e * NTOK;

    {
        int row  = tid >> 2;
        int cb   = (tid & 3) * 64;
        int gRow = tile0 + row;
        int tok  = (gRow < cntE) ? tl[gRow] : 0;
        const float* src = x + (size_t)tok * CDIM + cb;
        v8h* dst = (v8h*)(aS + row * CDIM + cb);
        #pragma unroll
        for (int i = 0; i < 64; i += 8) {
            v8h h;
            #pragma unroll
            for (int j = 0; j < 8; ++j) h[j] = (_Float16)src[i + j];
            dst[i >> 3] = h;
        }
    }

    v8f oacc[4][2];
    #pragma unroll
    for (int mm = 0; mm < 4; ++mm) { oacc[mm][0] = (v8f)0.0f; oacc[mm][1] = (v8f)0.0f; }

    for (int hc = 0; hc < HDIM / HC; ++hc) {
        const int hb = hc * HC;
        v8f hacc[4];
        #pragma unroll
        for (int mm = 0; mm < 4; ++mm) hacc[mm] = (v8f)0.0f;

        for (int k0 = 0; k0 < CDIM; k0 += 32) {
            __syncthreads();
            {
                int k  = tid >> 3;
                int nb = (tid & 7) * 16;
                const float* srow = Wfe + (size_t)(k0 + k) * HDIM + hb + nb;
                #pragma unroll
                for (int j = 0; j < 16; ++j)
                    bT[(nb + j) * 32 + k] = (_Float16)srow[j];
            }
            __syncthreads();
            v16h bfrag = *(const v16h*)(bT + (16 * w + col) * 32 + hiH * 16);
            #pragma unroll
            for (int mm = 0; mm < 4; ++mm) {
                const _Float16* ap = aS + (16 * mm + col) * CDIM + k0 + koff;
                v16h afrag;
                ((v8h*)&afrag)[0] = *(const v8h*)ap;
                ((v8h*)&afrag)[1] = *(const v8h*)(ap + 8);
                hacc[mm] = __builtin_amdgcn_wmma_f32_16x16x32_f16(
                    false, afrag, false, bfrag, (short)0, hacc[mm], false, false);
            }
        }

        #pragma unroll
        for (int mm = 0; mm < 4; ++mm)
            #pragma unroll
            for (int r = 0; r < 8; ++r) {
                float v = fmaxf(hacc[mm][r], 0.0f);
                v *= v;
                hS[(16 * mm + mofs + r) * HC + 16 * w + col] = (_Float16)v;
            }
        __syncthreads();

        for (int k0 = 0; k0 < HC; k0 += 32) {
            {
                int k  = tid >> 3;
                int nb = (tid & 7) * 32;
                const float* srow = Wpe + (size_t)(hb + k0 + k) * CDIM + nb;
                #pragma unroll
                for (int j = 0; j < 32; ++j)
                    bT[(nb + j) * 32 + k] = (_Float16)srow[j];
            }
            __syncthreads();
            v16h b0 = *(const v16h*)(bT + (32 * w + col) * 32 + hiH * 16);
            v16h b1 = *(const v16h*)(bT + (32 * w + 16 + col) * 32 + hiH * 16);
            #pragma unroll
            for (int mm = 0; mm < 4; ++mm) {
                const _Float16* ap = hS + (16 * mm + col) * HC + k0 + koff;
                v16h afrag;
                ((v8h*)&afrag)[0] = *(const v8h*)ap;
                ((v8h*)&afrag)[1] = *(const v8h*)(ap + 8);
                oacc[mm][0] = __builtin_amdgcn_wmma_f32_16x16x32_f16(
                    false, afrag, false, b0, (short)0, oacc[mm][0], false, false);
                oacc[mm][1] = __builtin_amdgcn_wmma_f32_16x16x32_f16(
                    false, afrag, false, b1, (short)0, oacc[mm][1], false, false);
            }
            __syncthreads();
        }
    }

    #pragma unroll
    for (int mm = 0; mm < 4; ++mm) {
        int Mbase = 16 * mm + mofs;
        #pragma unroll
        for (int r = 0; r < 8; ++r) {
            int gRow = tile0 + Mbase + r;
            if (gRow >= cntE) continue;
            int   tok = tl[gRow];
            float wt  = wl[gRow];
            float* orow = out + (size_t)tok * CDIM;
            atomicAdd(orow + 32 * w + col,      oacc[mm][0][r] * wt);
            atomicAdd(orow + 32 * w + 16 + col, oacc[mm][1][r] * wt);
        }
    }
}

// ---------------------------------------------------------------------------
extern "C" void kernel_launch(void* const* d_in, const int* in_sizes, int n_in,
                              void* d_out, int out_size, void* d_ws, size_t ws_size,
                              hipStream_t stream) {
    const float* x     = (const float*)d_in[0];
    const float* Wg    = (const float*)d_in[1];
    const float* Wfc   = (const float*)d_in[2];
    const float* Wpj   = (const float*)d_in[3];
    const float* ebias = (const float*)d_in[4];

    float* out = (float*)d_out;                    // [N, C]
    float* rw  = out + (size_t)NTOK * CDIM;        // [N, E]

    char*      ws    = (char*)d_ws;
    int*       cnt   = (int*)(ws + OFF_CNT);
    int*       tlist = (int*)(ws + OFF_TL);
    float*     wlist = (float*)(ws + OFF_WL);
    _Float16*  x16   = (_Float16*)(ws + OFF_X16);
    _Float16*  WfcT  = (_Float16*)(ws + OFF_WFC);
    _Float16*  WpjT  = (_Float16*)(ws + OFF_WPJ);

    const bool fast = (ws_size >= WS_NEED);

    moe_init<<<(NTOK * CDIM + 255) / 256, 256, 0, stream>>>(out, cnt);
    if (fast) {
        cvt_x<<<(NTOK * CDIM / 4 + 255) / 256, 256, 0, stream>>>(x, x16);
        dim3 gfc(HDIM / 32, CDIM / 32, EXPN);
        transpose_cvt<<<gfc, dim3(32, 8), 0, stream>>>(Wfc, WfcT, CDIM, HDIM);
        dim3 gpj(CDIM / 32, HDIM / 32, EXPN);
        transpose_cvt<<<gpj, dim3(32, 8), 0, stream>>>(Wpj, WpjT, HDIM, CDIM);
    }
    moe_gate<<<NTOK / 8, 256, 0, stream>>>(x, Wg, ebias, rw, cnt, tlist, wlist);

    dim3 grid(NTOK / TM, EXPN);
    if (fast)
        moe_expert_fast<<<grid, 256, 0, stream>>>(x16, WfcT, WpjT, cnt, tlist, wlist, out);
    else
        moe_expert_fb<<<grid, 256, 0, stream>>>(x, Wfc, Wpj, cnt, tlist, wlist, out);
}
